// SummaryAdapter_23648089932344
// MI455X (gfx1250) — matrix operand order repaired and checked by reading, not compiled, not run on hardware
//
#include <hip/hip_runtime.h>

// ---------------- shapes ----------------
#define LAY  8      // L
#define BB   2      // B
#define SS   4096   // S
#define DD   2048   // D (model dim)
#define NN   128    // N (bank slots)
#define DSUM 1024   // Ds (bank feature dim)
#define DA   64     // Da (adapter dim) == D_HEAD

typedef __attribute__((ext_vector_type(16))) __bf16 v16bf;
typedef __attribute__((ext_vector_type(8)))  float  v8f;

__device__ __forceinline__ v8f wmma_bf16(v16bf a, v16bf b, v8f c) {
  // D = A(16x32 bf16) * B(32x16 bf16) + C(16x16 f32)
  return __builtin_amdgcn_wmma_f32_16x16x32_bf16(
      /*neg_a=*/false, a, /*neg_b=*/false, b,
      /*c_mod=*/(short)0, c, /*reuse_a=*/false, /*reuse_b=*/false);
}

// Build a 16x32 bf16 A-fragment from f32 memory (global or LDS).
// p must already include this lane's K offset (lo = (lane&16)?16:0).
// A layout: lanes 0-15 hold K {0..15}; lanes 16-31 hold K {16..31}.
__device__ __forceinline__ v16bf a_frag_f32(const float* p) {
  const float4* q = (const float4*)p;
  float4 c0 = q[0], c1 = q[1];   // K +0..7  (relative)
  float4 c2 = q[2], c3 = q[3];   // K +8..15 (relative)
  v16bf a;
  a[0]=(__bf16)c0.x;  a[1]=(__bf16)c0.y;  a[2]=(__bf16)c0.z;  a[3]=(__bf16)c0.w;
  a[4]=(__bf16)c1.x;  a[5]=(__bf16)c1.y;  a[6]=(__bf16)c1.z;  a[7]=(__bf16)c1.w;
  a[8]=(__bf16)c2.x;  a[9]=(__bf16)c2.y;  a[10]=(__bf16)c2.z; a[11]=(__bf16)c2.w;
  a[12]=(__bf16)c3.x; a[13]=(__bf16)c3.y; a[14]=(__bf16)c3.z; a[15]=(__bf16)c3.w;
  return a;
}

// ---------------- prep kernels ----------------

// f32 -> bf16 copy (grid-stride)
__global__ void sa_cvt_bf16(const float* __restrict__ src,
                            __bf16* __restrict__ dst, int n) {
  int i = blockIdx.x * blockDim.x + threadIdx.x;
  int stride = gridDim.x * blockDim.x;
  for (; i < n; i += stride) dst[i] = (__bf16)src[i];
}

// K[l,b,n,a] = sum_c bank[b,n,c]*Wk[l,a,c]   -> Kb bf16, layout (l,b,N,Da)
// V[l,b,n,a] = sum_c bank[b,n,c]*Wv[l,a,c]   -> Vt bf16, layout (l,b,Da,N) (transposed)
__global__ void sa_kv_proj(const float* __restrict__ bank,
                           const float* __restrict__ Wk,
                           const float* __restrict__ Wv,
                           __bf16* __restrict__ Kb,
                           __bf16* __restrict__ Vt) {
  int idx = blockIdx.x * blockDim.x + threadIdx.x;     // over L*B*N*Da
  if (idx >= LAY * BB * NN * DA) return;
  int a = idx & (DA - 1);
  int n = (idx >> 6) & (NN - 1);
  int b = (idx >> 13) & (BB - 1);
  int l = idx >> 14;
  const float4* br = (const float4*)(bank + ((long)b * NN + n) * DSUM);
  const float4* wk = (const float4*)(Wk + ((long)l * DA + a) * DSUM);
  const float4* wv = (const float4*)(Wv + ((long)l * DA + a) * DSUM);
  float sk = 0.f, sv = 0.f;
  for (int c = 0; c < DSUM / 4; ++c) {
    float4 x = br[c], k = wk[c], v = wv[c];
    sk += x.x * k.x + x.y * k.y + x.z * k.z + x.w * k.w;
    sv += x.x * v.x + x.y * v.y + x.z * v.z + x.w * v.w;
  }
  Kb[(((long)l * BB + b) * NN + n) * DA + a] = (__bf16)sk;
  Vt[(((long)l * BB + b) * DA + a) * NN + n] = (__bf16)sv;
}

// ---------------- fused main kernel ----------------
// grid: L*B*(S/128) blocks, 256 threads (8 waves). Each wave owns 16 rows and a
// private 16x128 f32 LDS scratch -> no cross-wave synchronization needed.
__global__ __launch_bounds__(256) void sa_adapter_fused(
    const float* __restrict__ hidden,   // (L,B,S,D) f32
    const __bf16* __restrict__ Wqb,     // (L,Da,D)  bf16
    const __bf16* __restrict__ Wob,     // (L,D,Da)  bf16
    const __bf16* __restrict__ Kb,      // (L,B,N,Da) bf16
    const __bf16* __restrict__ Vt,      // (L,B,Da,N) bf16
    const float* __restrict__ gates,    // (L,)
    float* __restrict__ out) {          // (L,B,S,D) f32
  __shared__ float lds[8][16][128];     // 64 KB; per-wave 16x128 scratch

  const int tid  = threadIdx.x;
  const int wave = tid >> 5;
  const int lane = tid & 31;
  const int blk  = blockIdx.x;
  const int tile = blk & 31;            // S/128 tiles
  const int lb   = blk >> 5;
  const int b    = lb & 1;
  const int l    = lb >> 1;
  const int row0 = tile * 128 + wave * 16;

  const int mrow = lane & 15;           // A row / B column (lane&15)
  const int hi   = (lane >> 4) & 1;
  const int lo   = hi * 16;             // A-fragment K base
  const int kB   = hi * 16;             // B-fragment K base
  const int cm   = hi * 8;              // C-fragment row base
  const int cn   = mrow;                // C-fragment column

  const float* hbase = hidden + ((long)(l * BB + b)) * SS * DD;
  float (*Pw)[128] = lds[wave];

  // ---- Stage 1: Q = hidden * Wq^T  (16 x 64), scaled by 1/sqrt(Da) ----
  {
    v8f acc0 = {}, acc1 = {}, acc2 = {}, acc3 = {};
    const float* arow = hbase + (long)(row0 + mrow) * DD;
    const __bf16* wq = Wqb + (long)l * DA * DD;
    for (int kt = 0; kt < DD / 32; ++kt) {
      v16bf a = a_frag_f32(arow + kt * 32 + lo);
      const __bf16* b0 = wq + (long)cn * DD + kt * 32 + kB;
      acc0 = wmma_bf16(a, *(const v16bf*)(b0),           acc0);
      acc1 = wmma_bf16(a, *(const v16bf*)(b0 + 16 * DD), acc1);
      acc2 = wmma_bf16(a, *(const v16bf*)(b0 + 32 * DD), acc2);
      acc3 = wmma_bf16(a, *(const v16bf*)(b0 + 48 * DD), acc3);
    }
#pragma unroll
    for (int r = 0; r < 8; ++r) {
      Pw[cm + r][ 0 + cn] = acc0[r] * 0.125f;
      Pw[cm + r][16 + cn] = acc1[r] * 0.125f;
      Pw[cm + r][32 + cn] = acc2[r] * 0.125f;
      Pw[cm + r][48 + cn] = acc3[r] * 0.125f;
    }
  }

  // ---- Stage 2: scores = Q * K^T  (16 x 128) ----
  {
    v8f sc[8];
#pragma unroll
    for (int i = 0; i < 8; ++i) sc[i] = (v8f){};
    const __bf16* kmat = Kb + (long)(l * BB + b) * NN * DA;
#pragma unroll
    for (int kt = 0; kt < 2; ++kt) {
      v16bf a = a_frag_f32(&Pw[mrow][kt * 32 + lo]);
#pragma unroll
      for (int nt = 0; nt < 8; ++nt) {
        const __bf16* bp = kmat + (long)(nt * 16 + cn) * DA + kt * 32 + kB;
        sc[nt] = wmma_bf16(a, *(const v16bf*)bp, sc[nt]);
      }
    }
#pragma unroll
    for (int nt = 0; nt < 8; ++nt)
#pragma unroll
      for (int r = 0; r < 8; ++r)
        Pw[cm + r][nt * 16 + cn] = sc[nt][r];
  }

  // ---- Stage 3: softmax over N=128 per row (in place in LDS) ----
  {
    float* prow = &Pw[mrow][hi * 64];   // lane pair (lane, lane^16) split the row
    float mx = -3.0e38f;
    for (int j = 0; j < 64; j += 4) {
      float4 x = *(const float4*)(prow + j);
      mx = fmaxf(mx, fmaxf(fmaxf(x.x, x.y), fmaxf(x.z, x.w)));
    }
    mx = fmaxf(mx, __shfl_xor(mx, 16, 32));
    float sum = 0.f;
    for (int j = 0; j < 64; j += 4) {
      float4 x = *(const float4*)(prow + j);
      x.x = __expf(x.x - mx); x.y = __expf(x.y - mx);
      x.z = __expf(x.z - mx); x.w = __expf(x.w - mx);
      *(float4*)(prow + j) = x;
      sum += x.x + x.y + x.z + x.w;
    }
    sum += __shfl_xor(sum, 16, 32);
    float inv = 1.0f / sum;
    for (int j = 0; j < 64; j += 4) {
      float4 x = *(const float4*)(prow + j);
      x.x *= inv; x.y *= inv; x.z *= inv; x.w *= inv;
      *(float4*)(prow + j) = x;
    }
  }

  // ---- Stage 4: O = P * V  (16 x 64) ----
  {
    v8f oc0 = {}, oc1 = {}, oc2 = {}, oc3 = {};
    const __bf16* vt = Vt + (long)(l * BB + b) * DA * NN;
#pragma unroll
    for (int kt = 0; kt < 4; ++kt) {
      v16bf a = a_frag_f32(&Pw[mrow][kt * 32 + lo]);
      const __bf16* b0 = vt + (long)cn * NN + kt * 32 + kB;
      oc0 = wmma_bf16(a, *(const v16bf*)(b0),           oc0);
      oc1 = wmma_bf16(a, *(const v16bf*)(b0 + 16 * NN), oc1);
      oc2 = wmma_bf16(a, *(const v16bf*)(b0 + 32 * NN), oc2);
      oc3 = wmma_bf16(a, *(const v16bf*)(b0 + 48 * NN), oc3);
    }
#pragma unroll
    for (int r = 0; r < 8; ++r) {
      Pw[cm + r][ 0 + cn] = oc0[r];
      Pw[cm + r][16 + cn] = oc1[r];
      Pw[cm + r][32 + cn] = oc2[r];
      Pw[cm + r][48 + cn] = oc3[r];
    }
  }

  // ---- Stage 5: resid = O * Wo^T ; out = hidden + gate * resid ----
  {
    const float gate = 1.0f / (1.0f + __expf(-gates[l]));
    const __bf16* wo = Wob + (long)l * DD * DA;
    float* obase = out + ((long)(l * BB + b)) * SS * DD;
    for (int chunk = 0; chunk < DD / 64; ++chunk) {   // 32 chunks of 64 cols
      v8f rc0 = {}, rc1 = {}, rc2 = {}, rc3 = {};
#pragma unroll
      for (int kt = 0; kt < 2; ++kt) {
        v16bf a = a_frag_f32(&Pw[mrow][kt * 32 + lo]);
        const __bf16* b0 = wo + (long)(chunk * 64 + cn) * DA + kt * 32 + kB;
        rc0 = wmma_bf16(a, *(const v16bf*)(b0),           rc0);
        rc1 = wmma_bf16(a, *(const v16bf*)(b0 + 16 * DA), rc1);
        rc2 = wmma_bf16(a, *(const v16bf*)(b0 + 32 * DA), rc2);
        rc3 = wmma_bf16(a, *(const v16bf*)(b0 + 48 * DA), rc3);
      }
#pragma unroll
      for (int r = 0; r < 8; ++r) {
        long rowoff = (long)(row0 + cm + r) * DD + chunk * 64;
        obase[rowoff +  0 + cn] = hbase[rowoff +  0 + cn] + gate * rc0[r];
        obase[rowoff + 16 + cn] = hbase[rowoff + 16 + cn] + gate * rc1[r];
        obase[rowoff + 32 + cn] = hbase[rowoff + 32 + cn] + gate * rc2[r];
        obase[rowoff + 48 + cn] = hbase[rowoff + 48 + cn] + gate * rc3[r];
      }
    }
  }
}

// ---------------- launcher ----------------
extern "C" void kernel_launch(void* const* d_in, const int* in_sizes, int n_in,
                              void* d_out, int out_size, void* d_ws, size_t ws_size,
                              hipStream_t stream) {
  const float* hidden = (const float*)d_in[0];   // (L,B,S,D)
  const float* bank   = (const float*)d_in[1];   // (B,N,Ds)
  const float* Wq     = (const float*)d_in[2];   // (L,Da,D)
  const float* Wk     = (const float*)d_in[3];   // (L,Da,Ds)
  const float* Wv     = (const float*)d_in[4];   // (L,Da,Ds)
  const float* Wo     = (const float*)d_in[5];   // (L,D,Da)
  const float* gates  = (const float*)d_in[6];   // (L,)
  float* out = (float*)d_out;

  // workspace layout (bf16): Wq copy, Wo copy, K, V^T  (~4.7 MB total)
  char* ws = (char*)d_ws;
  size_t oWq = 0;
  size_t oWo = oWq + (size_t)LAY * DA * DD * 2;
  size_t oK  = oWo + (size_t)LAY * DD * DA * 2;
  size_t oV  = oK  + (size_t)LAY * BB * NN * DA * 2;
  __bf16* wqb = (__bf16*)(ws + oWq);
  __bf16* wob = (__bf16*)(ws + oWo);
  __bf16* kb  = (__bf16*)(ws + oK);
  __bf16* vt  = (__bf16*)(ws + oV);

  sa_cvt_bf16<<<512, 256, 0, stream>>>(Wq, wqb, LAY * DA * DD);
  sa_cvt_bf16<<<512, 256, 0, stream>>>(Wo, wob, LAY * DD * DA);
  sa_kv_proj<<<(LAY * BB * NN * DA + 255) / 256, 256, 0, stream>>>(bank, Wk, Wv, kb, vt);

  const int blocks = LAY * BB * (SS / 128);   // 512
  sa_adapter_fused<<<blocks, 256, 0, stream>>>(hidden, wqb, wob, kb, vt, gates, out);
}
